// TripletLoss_66666482368607
// MI455X (gfx1250) — matrix operand order repaired and checked by reading, not compiled, not run on hardware
//
#include <hip/hip_runtime.h>
#include <hip/hip_bf16.h>

typedef __attribute__((ext_vector_type(16))) _Float16 v16h;
typedef __attribute__((ext_vector_type(8)))  _Float16 v8h;
typedef __attribute__((ext_vector_type(8)))  float    v8f;

#define TL_EPS    1e-8f
#define TL_MARGIN 5.0f

union F16Frag { v16h v; v8h h[2]; };

// One fragment set for a K=32 step: shared A tile + 4 N-subtiles x 2 B operands.
struct Frags {
    F16Frag A;
    F16Frag B1[4];
    F16Frag B2[4];
};

__device__ __forceinline__ void load_frags(Frags& f,
                                           const _Float16* __restrict__ aptr,
                                           const _Float16* __restrict__ c1_0,
                                           const _Float16* __restrict__ c1_1,
                                           const _Float16* __restrict__ c1_2,
                                           const _Float16* __restrict__ c1_3,
                                           const _Float16* __restrict__ c2_0,
                                           const _Float16* __restrict__ c2_1,
                                           const _Float16* __restrict__ c2_2,
                                           const _Float16* __restrict__ c2_3,
                                           int k0) {
    f.A.h[0] = *(const v8h*)(aptr + k0);
    f.A.h[1] = *(const v8h*)(aptr + k0 + 8);
    const _Float16* p;
    p = c1_0 + k0; f.B1[0].h[0] = *(const v8h*)p; f.B1[0].h[1] = *(const v8h*)(p + 8);
    p = c1_1 + k0; f.B1[1].h[0] = *(const v8h*)p; f.B1[1].h[1] = *(const v8h*)(p + 8);
    p = c1_2 + k0; f.B1[2].h[0] = *(const v8h*)p; f.B1[2].h[1] = *(const v8h*)(p + 8);
    p = c1_3 + k0; f.B1[3].h[0] = *(const v8h*)p; f.B1[3].h[1] = *(const v8h*)(p + 8);
    p = c2_0 + k0; f.B2[0].h[0] = *(const v8h*)p; f.B2[0].h[1] = *(const v8h*)(p + 8);
    p = c2_1 + k0; f.B2[1].h[0] = *(const v8h*)p; f.B2[1].h[1] = *(const v8h*)(p + 8);
    p = c2_2 + k0; f.B2[2].h[0] = *(const v8h*)p; f.B2[2].h[1] = *(const v8h*)(p + 8);
    p = c2_3 + k0; f.B2[3].h[0] = *(const v8h*)p; f.B2[3].h[1] = *(const v8h*)(p + 8);
}

__device__ __forceinline__ void do_wmma(const Frags& f, v8f acc_aa[4], v8f acc_ag[4]) {
    #pragma unroll
    for (int t = 0; t < 4; ++t) {
        acc_aa[t] = __builtin_amdgcn_wmma_f32_16x16x32_f16(
            false, f.A.v, false, f.B1[t].v, (short)0, acc_aa[t], false, false);
        acc_ag[t] = __builtin_amdgcn_wmma_f32_16x16x32_f16(
            false, f.A.v, false, f.B2[t].v, (short)0, acc_ag[t], false, false);
    }
}

__global__ void tl_zero(double* __restrict__ acc) {
    if (threadIdx.x < 6) acc[threadIdx.x] = 0.0;
}

// One block per batch row: normalize anchor/positive/negative, emit fp32-exact
// d_ap/d_an contributions, write f16 normalized anchor & negative.
__global__ __launch_bounds__(256)
void tl_normalize(const float* __restrict__ feat,
                  _Float16* __restrict__ ha,
                  _Float16* __restrict__ hn,
                  double* __restrict__ acc,
                  int B, int D) {
    const int b   = blockIdx.x;
    const int tid = threadIdx.x;
    const float* arow = feat + ((size_t)b * 3 + 0) * D;
    const float* prow = feat + ((size_t)b * 3 + 1) * D;
    const float* nrow = feat + ((size_t)b * 3 + 2) * D;

    float sa = 0.f, sp = 0.f, sn = 0.f, dp = 0.f, dn = 0.f;
    for (int i = tid; i < D; i += blockDim.x) {
        float a = arow[i], p = prow[i], n = nrow[i];
        sa += a * a; sp += p * p; sn += n * n;
        dp += a * p; dn += a * n;
    }

    __shared__ float red[5][256];
    red[0][tid] = sa; red[1][tid] = sp; red[2][tid] = sn;
    red[3][tid] = dp; red[4][tid] = dn;
    __syncthreads();
    for (int s = 128; s > 0; s >>= 1) {
        if (tid < s) {
            #pragma unroll
            for (int c = 0; c < 5; ++c) red[c][tid] += red[c][tid + s];
        }
        __syncthreads();
    }

    __shared__ float ria_s, rin_s;
    if (tid == 0) {
        float ria = 1.0f / fmaxf(sqrtf(red[0][0]), TL_EPS);
        float rip = 1.0f / fmaxf(sqrtf(red[1][0]), TL_EPS);
        float rin = 1.0f / fmaxf(sqrtf(red[2][0]), TL_EPS);
        ria_s = ria; rin_s = rin;
        atomicAdd(acc + 4, (double)(1.0f - red[3][0] * ria * rip));  // d_ap
        atomicAdd(acc + 5, (double)(1.0f - red[4][0] * ria * rin));  // d_an
    }
    __syncthreads();
    const float ria = ria_s, rin = rin_s;
    for (int i = tid; i < D; i += blockDim.x) {
        ha[(size_t)b * D + i] = (_Float16)(arow[i] * ria);
        hn[(size_t)b * D + i] = (_Float16)(nrow[i] * rin);
    }
}

// Fused WMMA GEMM + masked upper-triangle reduction, software-pipelined.
// Block tile: 128 rows x 64 cols; 8 wave32s; wave w owns rows [i0+16w, +16).
__global__ __launch_bounds__(256)
void tl_pairs(const _Float16* __restrict__ ha,
              const _Float16* __restrict__ hn,
              const int* __restrict__ labels,
              const int* __restrict__ neg_labels,
              double* __restrict__ acc,
              int B, int D) {
    const int i0 = blockIdx.y * 128;
    const int j0 = blockIdx.x * 64;
    if (j0 + 63 <= i0) return;   // block-uniform: tile fully at/below diagonal

    const int tid  = threadIdx.x;
    const int wid  = tid >> 5;     // wave32
    const int lane = tid & 31;
    const int lhi  = lane >> 4;    // 0 or 1
    const int llo  = lane & 15;

    // A fragment (16x32 f16): lane holds row i0+16w+llo, 16 contiguous K
    // halves starting at k+16*lhi (matches B).
    const _Float16* aptr = ha + (size_t)(i0 + wid * 16 + llo) * D + lhi * 16;
    // B fragment (32x16 f16): lane holds col j0+16t+llo, 16 contiguous K
    // halves starting at k+16*lhi. B columns are rows of the source matrix.
    const int kB = lhi * 16;
    const size_t co0 = (size_t)(j0 + 0 * 16 + llo) * D + kB;
    const size_t co1 = (size_t)(j0 + 1 * 16 + llo) * D + kB;
    const size_t co2 = (size_t)(j0 + 2 * 16 + llo) * D + kB;
    const size_t co3 = (size_t)(j0 + 3 * 16 + llo) * D + kB;
    const _Float16* c1_0 = ha + co0; const _Float16* c2_0 = hn + co0;
    const _Float16* c1_1 = ha + co1; const _Float16* c2_1 = hn + co1;
    const _Float16* c1_2 = ha + co2; const _Float16* c2_2 = hn + co2;
    const _Float16* c1_3 = ha + co3; const _Float16* c2_3 = hn + co3;

    v8f acc_aa[4], acc_ag[4];
    const v8f z = {};
    #pragma unroll
    for (int t = 0; t < 4; ++t) { acc_aa[t] = z; acc_ag[t] = z; }

    // Double-buffered pipeline: while WMMAs consume one fragment set, the
    // 18 b128 loads of the next K half-step are in flight into the other.
    Frags f0, f1;
    load_frags(f0, aptr, c1_0, c1_1, c1_2, c1_3, c2_0, c2_1, c2_2, c2_3, 0);
    int k0 = 0;
    for (; k0 + 64 < D; k0 += 64) {
        load_frags(f1, aptr, c1_0, c1_1, c1_2, c1_3, c2_0, c2_1, c2_2, c2_3, k0 + 32);
        do_wmma(f0, acc_aa, acc_ag);
        load_frags(f0, aptr, c1_0, c1_1, c1_2, c1_3, c2_0, c2_1, c2_2, c2_3, k0 + 64);
        do_wmma(f1, acc_aa, acc_ag);
    }
    // Tail: last 64 K values (D is a multiple of 64).
    load_frags(f1, aptr, c1_0, c1_1, c1_2, c1_3, c2_0, c2_1, c2_2, c2_3, k0 + 32);
    do_wmma(f0, acc_aa, acc_ag);
    do_wmma(f1, acc_aa, acc_ag);

    // Epilogue: C layout -> element (M,N): lane = N + 16*(M/8), vgpr r = M%8.
    const int baseM = i0 + wid * 16 + lhi * 8;
    int lrow[8];
    #pragma unroll
    for (int r = 0; r < 8; ++r) lrow[r] = labels[baseM + r];

    float ps = 0.f, ns = 0.f, pc = 0.f, nc = 0.f;
    #pragma unroll
    for (int t = 0; t < 4; ++t) {
        const int gj = j0 + t * 16 + llo;
        const int lj = labels[gj];
        const int nj = neg_labels[gj];
        #pragma unroll
        for (int r = 0; r < 8; ++r) {
            const int gi = baseM + r;
            if (gj > gi) {                       // triu, k=1
                const float daa = 1.0f - acc_aa[t][r];
                const float dag = 1.0f - acc_ag[t][r];
                if (lrow[r] == lj) { ps += daa; pc += 1.f; }
                else               { ns += daa; nc += 1.f; }
                if (lrow[r] == nj) { ps += dag; pc += 1.f; }
                else               { ns += dag; nc += 1.f; }
            }
        }
    }

    __shared__ float red[4][256];
    red[0][tid] = ps; red[1][tid] = ns; red[2][tid] = pc; red[3][tid] = nc;
    __syncthreads();
    for (int s = 128; s > 0; s >>= 1) {
        if (tid < s) {
            red[0][tid] += red[0][tid + s];
            red[1][tid] += red[1][tid + s];
            red[2][tid] += red[2][tid + s];
            red[3][tid] += red[3][tid + s];
        }
        __syncthreads();
    }
    if (tid == 0) {
        atomicAdd(acc + 0, (double)red[0][0]);
        atomicAdd(acc + 1, (double)red[1][0]);
        atomicAdd(acc + 2, (double)red[2][0]);
        atomicAdd(acc + 3, (double)red[3][0]);
    }
}

__global__ void tl_final(const double* __restrict__ acc,
                         float* __restrict__ out, int B) {
    double pos_sum = acc[4] + acc[0];
    double neg_sum = acc[5] + acc[1];
    double pos_cnt = (double)B + acc[2];
    double neg_cnt = (double)B + acc[3];
    double loss = pos_sum / pos_cnt - neg_sum / neg_cnt + (double)TL_MARGIN;
    out[0] = (float)(loss > 0.0 ? loss : 0.0);
}

extern "C" void kernel_launch(void* const* d_in, const int* in_sizes, int n_in,
                              void* d_out, int out_size, void* d_ws, size_t ws_size,
                              hipStream_t stream) {
    const float* feat   = (const float*)d_in[0];
    const int*   labels = (const int*)d_in[1];
    const int*   neglab = (const int*)d_in[2];
    const int B = in_sizes[1];              // 4096
    const int D = in_sizes[0] / (3 * B);    // 1024

    double*   acc = (double*)d_ws;
    _Float16* ha  = (_Float16*)((char*)d_ws + 64);
    _Float16* hn  = ha + (size_t)B * D;
    float*    out = (float*)d_out;

    tl_zero<<<1, 32, 0, stream>>>(acc);
    tl_normalize<<<B, 256, 0, stream>>>(feat, ha, hn, acc, B, D);
    dim3 grid(B / 64, B / 128);             // 64 x 32 blocks
    tl_pairs<<<grid, 256, 0, stream>>>(ha, hn, labels, neglab, acc, B, D);
    tl_final<<<1, 1, 0, stream>>>(acc, out, B);
}
